// NodeEmbedder_22617297781140
// MI455X (gfx1250) — compile-verified, matrix operand order checked by reading, not run on hardware
//
#include <hip/hip_runtime.h>
#include <hip/hip_bf16.h>

#define NNODE 50000
#define NEDGE 800000
#define INDIM 768
#define HCD   128      // H*C
#define NH    4
#define CPH   32       // channels per head
#define OUTD  128
#define NEG_SLOPE 0.2f
#define MTILES 5       // M-tiles per wave; 50000/16 = 3125 = 5 * 625

typedef unsigned int   u32;
typedef unsigned short u16;
typedef __attribute__((ext_vector_type(16))) __bf16 v16bf;
typedef __attribute__((ext_vector_type(8)))  float  v8f;
typedef __attribute__((ext_vector_type(8)))  u32    v8u;

// ---------- helpers ----------
__device__ __forceinline__ u16 f2bf(float f) {
  u32 u = __builtin_bit_cast(u32, f);
  u = (u + 0x7FFFu + ((u >> 16) & 1u)) >> 16;   // round-to-nearest-even
  return (u16)u;
}
// monotonic float -> uint key (for atomicMax-based segment max); key(x)<key(y) iff x<y
__device__ __forceinline__ u32 fkey(float f) {
  u32 u = __builtin_bit_cast(u32, f);
  return (u & 0x80000000u) ? ~u : (u | 0x80000000u);
}
__device__ __forceinline__ float fdekey(u32 k) {
  u32 u = (k & 0x80000000u) ? (k & 0x7FFFFFFFu) : ~k;
  return __builtin_bit_cast(float, u);
}

// ---------- BatchNorm stats: one block per feature column ----------
__global__ void bn_stats_kernel(const float* __restrict__ x, int n, int din,
                                const float* __restrict__ g, const float* __restrict__ b,
                                float* __restrict__ scale, float* __restrict__ shift) {
  const int j = blockIdx.x;
  float s = 0.f, s2 = 0.f;
  for (int i = threadIdx.x; i < n; i += blockDim.x) {
    float v = x[(size_t)i * din + j];
    s += v; s2 += v * v;
  }
  __shared__ float sh[256], sh2[256];
  sh[threadIdx.x] = s; sh2[threadIdx.x] = s2;
  __syncthreads();
  for (int off = blockDim.x >> 1; off > 0; off >>= 1) {
    if (threadIdx.x < (unsigned)off) {
      sh[threadIdx.x]  += sh[threadIdx.x + off];
      sh2[threadIdx.x] += sh2[threadIdx.x + off];
    }
    __syncthreads();
  }
  if (threadIdx.x == 0) {
    float mean = sh[0] / (float)n;
    float var  = sh2[0] / (float)n - mean * mean;
    float sc   = g[j] * rsqrtf(var + 1e-5f);
    scale[j] = sc;
    shift[j] = b[j] - mean * sc;
  }
}

// ---------- normalize + fp32 -> bf16 ----------
__global__ void norm_to_bf16_kernel(const float* __restrict__ x,
                                    const float* __restrict__ scale,
                                    const float* __restrict__ shift,
                                    u16* __restrict__ out, int total, int din) {
  int i = blockIdx.x * blockDim.x + threadIdx.x;
  if (i >= total) return;
  int j = i % din;
  out[i] = f2bf(x[i] * scale[j] + shift[j]);
}

// ---------- fp32 -> bf16 ----------
__global__ void f32_to_bf16_kernel(const float* __restrict__ in, u16* __restrict__ out, int total) {
  int i = blockIdx.x * blockDim.x + threadIdx.x;
  if (i >= total) return;
  out[i] = f2bf(in[i]);
}

// ---------- weight convert + transpose: W[K][Ncols] -> Wt[Ncols][K] bf16 ----------
__global__ void wt_transpose_bf16_kernel(const float* __restrict__ W, u16* __restrict__ Wt,
                                         int K, int Ncols) {
  int i = blockIdx.x * blockDim.x + threadIdx.x;
  if (i >= K * Ncols) return;
  int k = i / Ncols, n = i % Ncols;
  Wt[(size_t)n * K + k] = f2bf(W[i]);
}

// ---------- bf16 WMMA GEMM: C[M x Ncols] = A[M x K] @ Bt[Ncols x K]^T (+bias, +relu) ----
// Each wave computes MTILES stacked 16x16 tiles (sharing one B fragment per k-step).
// grid.x = (M/16)/MTILES ; blockDim.x = 32 * (Ncols/16), Ncols <= 128.
// M/16 must be a multiple of MTILES (50000/16 = 3125 = 5*625): no tail, EXEC all-ones.
__global__ void gemm_bf16_wmma_kernel(const u16* __restrict__ A, const u16* __restrict__ Bt,
                                      float* __restrict__ C, int K, int ldc,
                                      const float* __restrict__ bias, int relu) {
  const int lane = threadIdx.x & 31;
  const int wave = threadIdx.x >> 5;
  const int half = lane >> 4;
  const int l16  = lane & 15;
  const int h8   = half * 8;
  const int rowTile0 = blockIdx.x * MTILES;

  const u16* Br = Bt + (size_t)(wave * 16 + l16) * K;      // output column's K-vector
  const u16* Ar[MTILES];
#pragma unroll
  for (int t = 0; t < MTILES; ++t)
    Ar[t] = A + (size_t)((rowTile0 + t) * 16 + l16) * K;   // A row for this lane, tile t

  v8f acc[MTILES];
#pragma unroll
  for (int t = 0; t < MTILES; ++t) acc[t] = (v8f){};

  for (int k0 = 0; k0 < K; k0 += 32) {
    // 16-bit fragment layout: k = base16(v) + half*8 + pairIndex*2 (bf16 pairs per VGPR)
    v8u bu;
#pragma unroll
    for (int v = 0; v < 8; ++v) {
      int k = k0 + ((v & 4) << 2) + h8 + ((v & 3) << 1);
      bu[v] = *(const u32*)(Br + k);
    }
    v16bf b = __builtin_bit_cast(v16bf, bu);
#pragma unroll
    for (int t = 0; t < MTILES; ++t) {
      v8u au;
#pragma unroll
      for (int v = 0; v < 8; ++v) {
        int k = k0 + ((v & 4) << 2) + h8 + ((v & 3) << 1);
        au[v] = *(const u32*)(Ar[t] + k);
      }
      v16bf a = __builtin_bit_cast(v16bf, au);
      acc[t] = __builtin_amdgcn_wmma_f32_16x16x32_bf16(false, a, false, b, (short)0,
                                                       acc[t], false, false);
    }
  }

  const int n  = wave * 16 + l16;
  const float bv = bias ? bias[n] : 0.0f;
#pragma unroll
  for (int t = 0; t < MTILES; ++t) {
    const int m0 = (rowTile0 + t) * 16 + h8;   // C layout: VGPR v -> row m0+v, lane -> col
#pragma unroll
    for (int v = 0; v < 8; ++v) {
      float val = acc[t][v] + bv;
      if (relu) val = val > 0.f ? val : 0.f;
      C[(size_t)(m0 + v) * ldc + n] = val;
    }
  }
}

// ---------- edge pass 1: logits + segment max (per edge,head) ----------
__global__ void edge_logits_kernel(const int* __restrict__ ei, const float* __restrict__ ea,
                                   const float* __restrict__ xl, const float* __restrict__ xr,
                                   const float* __restrict__ We, const float* __restrict__ att,
                                   float* __restrict__ logits, u32* __restrict__ lmaxk) {
  int t = blockIdx.x * blockDim.x + threadIdx.x;
  if (t >= NEDGE * NH) return;
  int e = t >> 2, h = t & 3;
  int s = ei[e], d = ei[NEDGE + e];
  const float* XL = xl + (size_t)s * HCD + h * CPH;
  const float* XR = xr + (size_t)d * HCD + h * CPH;
  float a0 = ea[e * 3], a1 = ea[e * 3 + 1], a2 = ea[e * 3 + 2];
  float logit = 0.f;
#pragma unroll 8
  for (int c = 0; c < CPH; ++c) {
    int jc = h * CPH + c;
    float ev = a0 * We[jc] + a1 * We[HCD + jc] + a2 * We[2 * HCD + jc];
    float m = XL[c] + XR[c] + ev;
    m = m > 0.f ? m : NEG_SLOPE * m;
    logit += m * att[jc];
  }
  logits[t] = logit;
  atomicMax(&lmaxk[d * NH + h], fkey(logit));
}

// ---------- edge pass 2: exp + segment sum ----------
__global__ void edge_exp_kernel(const int* __restrict__ ei,
                                const float* __restrict__ logits,
                                const u32* __restrict__ lmaxk,
                                float* __restrict__ exb, float* __restrict__ den) {
  int t = blockIdx.x * blockDim.x + threadIdx.x;
  if (t >= NEDGE * NH) return;
  int e = t >> 2, h = t & 3;
  int d = ei[NEDGE + e];
  float ex = __expf(logits[t] - fdekey(lmaxk[d * NH + h]));
  exb[t] = ex;
  atomicAdd(&den[d * NH + h], ex);
}

// ---------- edge pass 3: weighted scatter-add aggregation ----------
__global__ void edge_agg_kernel(const int* __restrict__ ei,
                                const float* __restrict__ xl,
                                const float* __restrict__ exb, const float* __restrict__ den,
                                float* __restrict__ agg) {
  int t = blockIdx.x * blockDim.x + threadIdx.x;
  if (t >= NEDGE * NH) return;
  int e = t >> 2, h = t & 3;
  int s = ei[e], d = ei[NEDGE + e];
  float alpha = exb[t] / (den[d * NH + h] + 1e-16f);
  const float* XL = xl + (size_t)s * HCD + h * CPH;
  float* AG = agg + (size_t)d * HCD + h * CPH;
#pragma unroll 8
  for (int c = 0; c < CPH; ++c) atomicAdd(&AG[c], XL[c] * alpha);
}

// ---------- agg + conv_b -> bf16 (input to lin GEMM) ----------
__global__ void add_bias_to_bf16_kernel(const float* __restrict__ agg,
                                        const float* __restrict__ cb,
                                        u16* __restrict__ out, int total) {
  int i = blockIdx.x * blockDim.x + threadIdx.x;
  if (i >= total) return;
  out[i] = f2bf(agg[i] + cb[i & (HCD - 1)]);
}

// ==========================================================================
extern "C" void kernel_launch(void* const* d_in, const int* in_sizes, int n_in,
                              void* d_out, int out_size, void* d_ws, size_t ws_size,
                              hipStream_t stream) {
  (void)in_sizes; (void)n_in; (void)out_size; (void)ws_size;

  const float* x   = (const float*)d_in[0];
  const int*   ei  = (const int*)d_in[1];
  const float* ea  = (const float*)d_in[2];
  // per-layer params: bn_g, bn_b, Wl, Wr, We, att, conv_b, lin_W, lin_b
  const float* LP[2][9];
  for (int l = 0; l < 2; ++l)
    for (int k = 0; k < 9; ++k)
      LP[l][k] = (const float*)d_in[3 + l * 9 + k];
  const float* out_W1 = (const float*)d_in[21];
  const float* out_b1 = (const float*)d_in[22];
  const float* out_W2 = (const float*)d_in[23];
  const float* out_b2 = (const float*)d_in[24];
  float* yout = (float*)d_out;

  // ---- workspace arena ----
  char* w = (char*)d_ws;
  auto alloc = [&](size_t bytes) -> void* {
    void* p = (void*)w;
    w += (bytes + 255) & ~(size_t)255;
    return p;
  };
  u16*   hbf    = (u16*)  alloc((size_t)NNODE * INDIM * 2);  // bf16 A-matrix (reused)
  u16*   wlt    = (u16*)  alloc((size_t)INDIM * HCD * 2);
  u16*   wrt    = (u16*)  alloc((size_t)INDIM * HCD * 2);
  u16*   lint   = (u16*)  alloc((size_t)HCD * HCD * 2);
  u16*   w1t    = (u16*)  alloc((size_t)HCD * CPH * 2);
  u16*   w2t    = (u16*)  alloc((size_t)CPH * OUTD * 2);
  float* xl     = (float*)alloc((size_t)NNODE * HCD * 4);
  float* xr     = (float*)alloc((size_t)NNODE * HCD * 4);
  float* logits = (float*)alloc((size_t)NEDGE * NH * 4);
  float* exb    = (float*)alloc((size_t)NEDGE * NH * 4);
  u32*   lmaxk  = (u32*)  alloc((size_t)NNODE * NH * 4);
  float* den    = (float*)alloc((size_t)NNODE * NH * 4);
  float* agg    = (float*)alloc((size_t)NNODE * HCD * 4);
  float* hcur   = (float*)alloc((size_t)NNODE * HCD * 4);
  float* out1   = (float*)alloc((size_t)NNODE * CPH * 4);
  float* scale  = (float*)alloc((size_t)INDIM * 4);
  float* shift  = (float*)alloc((size_t)INDIM * 4);

  const int GT = (NNODE / 16) / MTILES;            // 625 blocks; 5 M-tiles per wave
  const int EH = NEDGE * NH;
  const int egrid = (EH + 255) / 256;

  for (int l = 0; l < 2; ++l) {
    const int din = (l == 0) ? INDIM : HCD;
    const float* hin = (l == 0) ? x : hcur;

    // BatchNorm fold: scale/shift per column
    bn_stats_kernel<<<din, 256, 0, stream>>>(hin, NNODE, din, LP[l][0], LP[l][1], scale, shift);
    {
      int tot = NNODE * din;
      norm_to_bf16_kernel<<<(tot + 255) / 256, 256, 0, stream>>>(hin, scale, shift, hbf, tot, din);
    }

    // weights -> bf16 transposed
    wt_transpose_bf16_kernel<<<(din * HCD + 255) / 256, 256, 0, stream>>>(LP[l][2], wlt, din, HCD);
    wt_transpose_bf16_kernel<<<(din * HCD + 255) / 256, 256, 0, stream>>>(LP[l][3], wrt, din, HCD);
    wt_transpose_bf16_kernel<<<(HCD * HCD + 255) / 256, 256, 0, stream>>>(LP[l][7], lint, HCD, HCD);

    // xl = h @ Wl, xr = h @ Wr   (WMMA bf16, fp32 accumulate)
    gemm_bf16_wmma_kernel<<<GT, 256, 0, stream>>>(hbf, wlt, xl, din, HCD, nullptr, 0);
    gemm_bf16_wmma_kernel<<<GT, 256, 0, stream>>>(hbf, wrt, xr, din, HCD, nullptr, 0);

    // reset segment buffers every launch (harness does not re-poison)
    hipMemsetAsync(lmaxk, 0, (size_t)NNODE * NH * 4, stream);
    hipMemsetAsync(den,   0, (size_t)NNODE * NH * 4, stream);
    hipMemsetAsync(agg,   0, (size_t)NNODE * HCD * 4, stream);

    edge_logits_kernel<<<egrid, 256, 0, stream>>>(ei, ea, xl, xr, LP[l][4], LP[l][5],
                                                  logits, lmaxk);
    edge_exp_kernel<<<egrid, 256, 0, stream>>>(ei, logits, lmaxk, exb, den);
    edge_agg_kernel<<<egrid, 256, 0, stream>>>(ei, xl, exb, den, agg);

    // h = relu((agg + conv_b) @ lin_W + lin_b)
    {
      int tot = NNODE * HCD;
      add_bias_to_bf16_kernel<<<(tot + 255) / 256, 256, 0, stream>>>(agg, LP[l][6], hbf, tot);
    }
    gemm_bf16_wmma_kernel<<<GT, 256, 0, stream>>>(hbf, lint, hcur, HCD, HCD, LP[l][8], 1);
  }

  // output head: out1 = h @ out_W1 + b1 ; y = out1 @ out_W2 + b2
  wt_transpose_bf16_kernel<<<(HCD * CPH + 255) / 256, 256, 0, stream>>>(out_W1, w1t, HCD, CPH);
  wt_transpose_bf16_kernel<<<(CPH * OUTD + 255) / 256, 256, 0, stream>>>(out_W2, w2t, CPH, OUTD);

  {
    int tot = NNODE * HCD;
    f32_to_bf16_kernel<<<(tot + 255) / 256, 256, 0, stream>>>(hcur, hbf, tot);
  }
  gemm_bf16_wmma_kernel<<<GT, 64, 0, stream>>>(hbf, w1t, out1, HCD, CPH, out_b1, 0);

  {
    int tot = NNODE * CPH;
    f32_to_bf16_kernel<<<(tot + 255) / 256, 256, 0, stream>>>(out1, hbf, tot);
  }
  gemm_bf16_wmma_kernel<<<GT, 256, 0, stream>>>(hbf, w2t, yout, CPH, OUTD, out_b2, 0);
}
